// S6_6828998001058
// MI455X (gfx1250) — matrix operand order repaired and checked by reading, not compiled, not run on hardware
//
#include <hip/hip_runtime.h>
#include <hip/hip_bf16.h>
#include <math.h>

// ---------------- problem constants (from reference) ----------------
#define D_MODEL 2048
#define D_STATE 16
#define DT_RANK 128
#define B_SZ    4
#define SEQLEN  2048
#define M_TOT   (B_SZ * SEQLEN)          // 8192 rows (b*L)
#define N_PROJ  (DT_RANK + 2 * D_STATE)  // 160

// ---------------- CDNA5 WMMA types ----------------
typedef __attribute__((ext_vector_type(16))) __bf16 v16bf;
typedef __attribute__((ext_vector_type(8)))  __bf16 v8bf;
typedef __attribute__((ext_vector_type(8)))  float  v8f;

static __device__ __forceinline__ unsigned short f2bf(float f) {
    unsigned int u = __float_as_uint(f);
    unsigned int r = ((u >> 16) & 1u) + 0x7FFFu;   // round-to-nearest-even
    return (unsigned short)((u + r) >> 16);
}

union FragU { v16bf v; v8bf h[2]; };

// A-fragment (16x32 bf16, row-major bf16 A, stride lda):
// lanes 0-15: M=lane,   K = {k..k+7,  k+16..k+23}
// lanes16-31: M=lane-16,K = {k+8..15, k+24..31}
static __device__ __forceinline__ v16bf load_a_frag(const __bf16* A, int lda,
                                                    int m0, int k, int lane) {
    int m  = m0 + (lane & 15);
    int kk = k + ((lane & 16) ? 8 : 0);
    const __bf16* p = A + (size_t)m * lda + kk;
    FragU u;
    u.h[0] = *reinterpret_cast<const v8bf*>(p);        // 16B contiguous
    u.h[1] = *reinterpret_cast<const v8bf*>(p + 16);   // 16B contiguous
    return u.v;
}

// A-fragment built from row-major *f32* A, converting to bf16 in-register
// (fuses the downconvert into the GEMM; A is read exactly once by GEMM1).
static __device__ __forceinline__ v16bf load_a_frag_f32(const float* A, int lda,
                                                        int m0, int k, int lane) {
    int m  = m0 + (lane & 15);
    int kk = k + ((lane & 16) ? 8 : 0);
    const float* p = A + (size_t)m * lda + kk;
    v8f lo = *reinterpret_cast<const v8f*>(p);         // 32B contiguous
    v8f hi = *reinterpret_cast<const v8f*>(p + 16);    // 32B contiguous
    v16bf r;
#pragma unroll
    for (int i = 0; i < 8; ++i) {
        r[i]     = (__bf16)lo[i];
        r[i + 8] = (__bf16)hi[i];
    }
    return r;
}

// B-fragment (32x16 bf16) from weights stored (N,K) row-major (PyTorch layout):
// lanes 0-15: N=lane,   K = {k..k+7,  k+16..k+23}
// lanes16-31: N=lane-16,K = {k+8..15, k+24..31}
static __device__ __forceinline__ v16bf load_b_frag(const __bf16* Bt, int ldb,
                                                    int n0, int k, int lane) {
    int n  = n0 + (lane & 15);
    int kk = k + ((lane & 16) ? 8 : 0);
    const __bf16* p = Bt + (size_t)n * ldb + kk;
    FragU u;
    u.h[0] = *reinterpret_cast<const v8bf*>(p);
    u.h[1] = *reinterpret_cast<const v8bf*>(p + 16);
    return u.v;
}

#define WMMA_BF16(A_, B_, C_) \
    __builtin_amdgcn_wmma_f32_16x16x32_bf16(false, (A_), false, (B_), (short)0, (C_), false, false)

// ---------------- elementwise f32 -> bf16 (weights only) ----------------
__global__ void k_f32_to_bf16(const float* __restrict__ src,
                              unsigned short* __restrict__ dst, int n) {
    for (int i = blockIdx.x * blockDim.x + threadIdx.x; i < n;
         i += gridDim.x * blockDim.x)
        dst[i] = f2bf(src[i]);
}

// ---------------- GEMM1: fused input projection ----------------
// [dt_low | Bm | Cm](8192x160) = hidden(8192x2048, f32) @ x_proj_weight^T
// 5 waves/block; each wave owns a 64x32 tile -> 64x160 block tile.
// Waves 0-3 produce dt_low columns; wave 4 produces Bm (c=0) and Cm (c=1):
// the epilogue split is wave-uniform -> no divergent stores.
__global__ __launch_bounds__(160) void k_proj_gemm(
    const float* __restrict__ hid,
    const unsigned short* __restrict__ xpw_bf,
    unsigned short* __restrict__ dtlow_bf,
    float* __restrict__ Bm, float* __restrict__ Cm) {
    const __bf16* Bt = reinterpret_cast<const __bf16*>(xpw_bf);
    const int lane = threadIdx.x & 31;
    const int w    = threadIdx.x >> 5;       // 0..4
    const int m0   = blockIdx.x * 64;
    const int n0   = w * 32;
    v8f acc[4][2] = {};
    for (int k = 0; k < D_MODEL; k += 32) {
        __builtin_prefetch(hid + (size_t)m0 * D_MODEL + k + 256, 0, 1);
        v16bf av[4];
#pragma unroll
        for (int r = 0; r < 4; ++r)
            av[r] = load_a_frag_f32(hid, D_MODEL, m0 + 16 * r, k, lane);
#pragma unroll
        for (int c = 0; c < 2; ++c) {
            v16bf bv = load_b_frag(Bt, D_MODEL, n0 + 16 * c, k, lane);
#pragma unroll
            for (int r = 0; r < 4; ++r)
                acc[r][c] = WMMA_BF16(av[r], bv, acc[r][c]);
        }
    }
    const int rofs = (lane & 16) ? 8 : 0;
    const int cofs = lane & 15;
    if (w < 4) {  // uniform branch: dt_low columns 0..127
        for (int r = 0; r < 4; ++r)
            for (int c = 0; c < 2; ++c)
                for (int v = 0; v < 8; ++v) {
                    int row = m0 + 16 * r + v + rofs;
                    int col = n0 + 16 * c + cofs;
                    dtlow_bf[(size_t)row * DT_RANK + col] = f2bf(acc[r][c][v]);
                }
    } else {      // wave 4: c==0 -> Bm, c==1 -> Cm (compile-time split)
        for (int r = 0; r < 4; ++r)
            for (int v = 0; v < 8; ++v) {
                int row = m0 + 16 * r + v + rofs;
                Bm[(size_t)row * D_STATE + cofs] = acc[r][0][v];
                Cm[(size_t)row * D_STATE + cofs] = acc[r][1][v];
            }
    }
}

static __device__ __forceinline__ float softplus_f(float x) {
    return fmaxf(x, 0.0f) + log1pf(__expf(-fabsf(x)));
}

// ---------------- GEMM2: delta = softplus(dt_low @ dt_proj^T + bias) ----------------
// M=8192, N=2048, K=128.  8 waves, 64x64 wave tile -> 128x256 block tile.
__global__ __launch_bounds__(256) void k_dt_gemm(
    const unsigned short* __restrict__ dtlow_bf,
    const unsigned short* __restrict__ dtw_bf,
    const float* __restrict__ dt_bias,
    float* __restrict__ delta) {
    const __bf16* A  = reinterpret_cast<const __bf16*>(dtlow_bf);
    const __bf16* Bt = reinterpret_cast<const __bf16*>(dtw_bf);
    const int lane = threadIdx.x & 31;
    const int w    = threadIdx.x >> 5;
    const int m0   = blockIdx.y * 128 + (w >> 2) * 64;
    const int n0   = blockIdx.x * 256 + (w & 3) * 64;
    v8f acc[4][4] = {};
    for (int k = 0; k < DT_RANK; k += 32) {
        v16bf av[4];
#pragma unroll
        for (int r = 0; r < 4; ++r)
            av[r] = load_a_frag(A, DT_RANK, m0 + 16 * r, k, lane);
#pragma unroll
        for (int c = 0; c < 4; ++c) {
            v16bf bv = load_b_frag(Bt, DT_RANK, n0 + 16 * c, k, lane);
#pragma unroll
            for (int r = 0; r < 4; ++r)
                acc[r][c] = WMMA_BF16(av[r], bv, acc[r][c]);
        }
    }
    const int rofs = (lane & 16) ? 8 : 0;
    const int cofs = lane & 15;
    for (int c = 0; c < 4; ++c) {
        int col = n0 + 16 * c + cofs;
        float bias = dt_bias[col];
        for (int r = 0; r < 4; ++r)
            for (int v = 0; v < 8; ++v) {
                int row = m0 + 16 * r + v + rofs;
                delta[(size_t)row * D_MODEL + col] =
                    softplus_f(acc[r][c][v] + bias);
            }
    }
}

// ---------------- Selective scan ----------------
// One thread per (b,d) channel, 16-state recurrence in registers.
// B_t/C_t tiles staged memory->LDS with GLOBAL_LOAD_ASYNC_TO_LDS_B128
// (ASYNCcnt path), one b128 per thread per tile.
#define T_TILE 64
__global__ __launch_bounds__(256) void k_scan(
    const float* __restrict__ delta, const float* __restrict__ Bm,
    const float* __restrict__ Cm, const float* __restrict__ hid,
    const float* __restrict__ A_log, unsigned short* __restrict__ ybf) {
    __shared__ __align__(16) float sB[T_TILE * D_STATE];   // 4 KB
    __shared__ __align__(16) float sC[T_TILE * D_STATE];   // 4 KB
    const int b = blockIdx.x >> 3;
    const int d = (blockIdx.x & 7) * 256 + threadIdx.x;
    // LDS byte offsets for this thread's 16B slice (generic ptr low 32 bits
    // are the LDS offset per the aperture rules).
    const unsigned ldsB =
        (unsigned)(unsigned long long)(void*)sB + (unsigned)threadIdx.x * 16u;
    const unsigned ldsC =
        (unsigned)(unsigned long long)(void*)sC + (unsigned)threadIdx.x * 16u;
    float a[D_STATE], h[D_STATE];
#pragma unroll
    for (int n = 0; n < D_STATE; ++n) {
        a[n] = -__expf(A_log[d * D_STATE + n]);   // A = -exp(A_log)
        h[n] = 0.0f;
    }
    for (int t0 = 0; t0 < SEQLEN; t0 += T_TILE) {
        const float* gB =
            Bm + ((size_t)b * SEQLEN + t0) * D_STATE + threadIdx.x * 4;
        const float* gC =
            Cm + ((size_t)b * SEQLEN + t0) * D_STATE + threadIdx.x * 4;
        // async memory -> LDS staging (no VGPR round trip, ASYNCcnt tracked)
        asm volatile("global_load_async_to_lds_b128 %0, %1, off"
                     :: "v"(ldsB), "v"(gB) : "memory");
        asm volatile("global_load_async_to_lds_b128 %0, %1, off"
                     :: "v"(ldsC), "v"(gC) : "memory");
        asm volatile("s_wait_asynccnt 0" ::: "memory");
        __syncthreads();
        for (int tt = 0; tt < T_TILE; ++tt) {
            size_t row = (size_t)b * SEQLEN + t0 + tt;
            float dlt = delta[row * D_MODEL + d];
            float x   = hid[row * D_MODEL + d];
            float dx  = dlt * x;
            float y   = 0.0f;
#pragma unroll
            for (int n = 0; n < D_STATE; ++n) {
                float dA = __expf(dlt * a[n]);
                h[n] = fmaf(h[n], dA, dx * sB[tt * D_STATE + n]);
                y    = fmaf(h[n], sC[tt * D_STATE + n], y);
            }
            ybf[row * D_MODEL + d] = f2bf(y);
        }
        __syncthreads();
    }
}

// ---------------- GEMM3: out = relu(y @ out_proj^T + b) + D*hidden ----------------
// M=8192, N=2048, K=2048.  64x64 wave tile: 16 b128 loads feed 16 WMMAs.
__global__ __launch_bounds__(256) void k_out_gemm(
    const unsigned short* __restrict__ ybf,
    const unsigned short* __restrict__ outw_bf,
    const float* __restrict__ out_bias, const float* __restrict__ Dv,
    const float* __restrict__ hid, float* __restrict__ out) {
    const __bf16* A  = reinterpret_cast<const __bf16*>(ybf);
    const __bf16* Bt = reinterpret_cast<const __bf16*>(outw_bf);
    const int lane = threadIdx.x & 31;
    const int w    = threadIdx.x >> 5;
    const int m0   = blockIdx.y * 128 + (w >> 2) * 64;
    const int n0   = blockIdx.x * 256 + (w & 3) * 64;
    v8f acc[4][4] = {};
    for (int k = 0; k < D_MODEL; k += 32) {
        __builtin_prefetch(A + (size_t)m0 * D_MODEL + k + 256, 0, 1);
        v16bf av[4];
#pragma unroll
        for (int r = 0; r < 4; ++r)
            av[r] = load_a_frag(A, D_MODEL, m0 + 16 * r, k, lane);
#pragma unroll
        for (int c = 0; c < 4; ++c) {
            v16bf bv = load_b_frag(Bt, D_MODEL, n0 + 16 * c, k, lane);
#pragma unroll
            for (int r = 0; r < 4; ++r)
                acc[r][c] = WMMA_BF16(av[r], bv, acc[r][c]);
        }
    }
    const int rofs = (lane & 16) ? 8 : 0;
    const int cofs = lane & 15;
    for (int c = 0; c < 4; ++c) {
        int col = n0 + 16 * c + cofs;
        float bias = out_bias[col];
        float dsc  = Dv[col];
        for (int r = 0; r < 4; ++r)
            for (int v = 0; v < 8; ++v) {
                int row = m0 + 16 * r + v + rofs;
                float val = fmaxf(acc[r][c][v] + bias, 0.0f);
                val += dsc * hid[(size_t)row * D_MODEL + col];
                out[(size_t)row * D_MODEL + col] = val;
            }
    }
}

// ---------------- launcher ----------------
extern "C" void kernel_launch(void* const* d_in, const int* in_sizes, int n_in,
                              void* d_out, int out_size, void* d_ws,
                              size_t ws_size, hipStream_t stream) {
    (void)in_sizes; (void)n_in; (void)out_size; (void)ws_size;
    const float* hid   = (const float*)d_in[0];  // (4,2048,2048)
    const float* xpw   = (const float*)d_in[1];  // (160,2048)
    const float* dtw   = (const float*)d_in[2];  // (2048,128)
    const float* dtb   = (const float*)d_in[3];  // (2048,)
    const float* A_log = (const float*)d_in[4];  // (2048,16)
    const float* Dv    = (const float*)d_in[5];  // (2048,)
    const float* outw  = (const float*)d_in[6];  // (2048,2048)
    const float* outb  = (const float*)d_in[7];  // (2048,)
    float* out = (float*)d_out;

    char*  ws  = (char*)d_ws;
    size_t off = 0;
    auto carve = [&](size_t bytes) -> char* {
        char* p = ws + off;
        off = (off + bytes + 255) & ~(size_t)255;
        return p;
    };
    unsigned short* xpw_bf   = (unsigned short*)carve((size_t)N_PROJ * D_MODEL * 2);
    unsigned short* dtw_bf   = (unsigned short*)carve((size_t)D_MODEL * DT_RANK * 2);
    unsigned short* outw_bf  = (unsigned short*)carve((size_t)D_MODEL * D_MODEL * 2);
    unsigned short* dtlow_bf = (unsigned short*)carve((size_t)M_TOT * DT_RANK * 2);
    float* Bm    = (float*)carve((size_t)M_TOT * D_STATE * 4);
    float* Cm    = (float*)carve((size_t)M_TOT * D_STATE * 4);
    float* delta = (float*)carve((size_t)M_TOT * D_MODEL * 4);
    unsigned short* ybf = (unsigned short*)carve((size_t)M_TOT * D_MODEL * 2);

    // weights -> bf16 (reused many times from L2; activations convert in-GEMM)
    k_f32_to_bf16<<<320,  256, 0, stream>>>(xpw,  xpw_bf,  N_PROJ * D_MODEL);
    k_f32_to_bf16<<<256,  256, 0, stream>>>(dtw,  dtw_bf,  D_MODEL * DT_RANK);
    k_f32_to_bf16<<<4096, 256, 0, stream>>>(outw, outw_bf, D_MODEL * D_MODEL);

    // fused input projections (dt_low | B | C) with in-loop f32->bf16 of A
    k_proj_gemm<<<dim3(M_TOT / 64), 160, 0, stream>>>(hid, xpw_bf, dtlow_bf,
                                                      Bm, Cm);
    // delta = softplus(dt_low @ dt_proj^T + bias)  (rank-128 factorization)
    k_dt_gemm<<<dim3(D_MODEL / 256, M_TOT / 128), 256, 0, stream>>>(
        dtlow_bf, dtw_bf, dtb, delta);
    // sequential selective scan -> y (bf16)
    k_scan<<<dim3(B_SZ * (D_MODEL / 256)), 256, 0, stream>>>(delta, Bm, Cm, hid,
                                                             A_log, ybf);
    // out = relu(y @ out_proj^T + bias) + D * hidden
    k_out_gemm<<<dim3(D_MODEL / 256, M_TOT / 128), 256, 0, stream>>>(
        ybf, outw_bf, outb, Dv, hid, out);
}
